// Transformer_61959198212425
// MI455X (gfx1250) — matrix operand order repaired and checked by reading, compiled, not run on hardware
//
#include <hip/hip_runtime.h>

// ---------------------------------------------------------------------------
// MI455X (gfx1250) transformer forward. All GEMMs run on v_wmma_f32_16x16x32_bf16
// (wave32, 16x16 tiles, fp32 accumulate). Weights are converted to transposed
// bf16 once per launch so every WMMA B-fragment is a contiguous 16B load.
// GEMM inner loop is a ping-pong (unroll-2) software pipeline with the final
// k-chunk peeled, so fragment buffers rotate by position (no register copies)
// and the 8 WMMAs of each half-step cover the other half-step's load latency.
// ---------------------------------------------------------------------------

typedef __bf16 bf16;
typedef __attribute__((ext_vector_type(16))) bf16  v16bf;
typedef __attribute__((ext_vector_type(8)))  bf16  v8bf;
typedef __attribute__((ext_vector_type(8)))  float v8f;

// model constants
static constexpr int Dm  = 512;     // model dim
static constexpr int Hn  = 8;       // heads
static constexpr int DHd = 64;      // head dim
static constexpr int Fff = 2048;    // ffn dim
static constexpr int LEn = 6;       // encoder layers
static constexpr int LDn = 6;       // decoder layers
static constexpr int Vv  = 32000;   // vocab
static constexpr int Bn  = 4;       // batch
static constexpr int Tn  = 512;     // seq len (src == tgt)
static constexpr int NT  = Bn * Tn; // 2048 token rows

// ---------------------------------------------------------------------------
// Fragment loaders. Contiguous k order used by this code, same for A and B (16x32 A / 32x16 B, wave32):
//   A: lane l (g=l/16, m=l%16), elem e -> K = 16g+e,                       row m
//   B: lane l (g=l/16, n=l%16), elem e -> K = 16g+e,                       col n
// Both reduce to contiguous 16B loads from K-major storage.
// ---------------------------------------------------------------------------
__device__ inline v16bf ld_afrag(const bf16* p) {
  union { v16bf v; v8bf h[2]; } u;
  u.h[0] = *(const v8bf*)(p);        // K = k0+16g .. +7
  u.h[1] = *(const v8bf*)(p + 8);    // K = k0+16g+8 .. +15
  return u.v;
}
__device__ inline v16bf ld_bfrag(const bf16* p) {
  union { v16bf v; v8bf h[2]; } u;
  u.h[0] = *(const v8bf*)(p);        // K = k0+16g .. +7
  u.h[1] = *(const v8bf*)(p + 8);    // K = k0+16g+8 .. +15
  return u.v;
}
__device__ inline v8f wmma_bf16(v16bf a, v16bf b, v8f c) {
  return __builtin_amdgcn_wmma_f32_16x16x32_bf16(
      false, a, false, b, (short)0, c, false, false);
}

// ---------------------------------------------------------------------------
// NT GEMM: C[M,N] (fp32) = A[M,K](bf16, row-major) * Bt[N,K](bf16)^T
// One wave -> 32x64 tile: 2 A-frags x 4 B-frags -> 8 WMMAs per 32-wide k-step
// (6 load-pairs / 8 WMMAs). Block = 128 thr = 4 waves -> 128x64 per block.
// Requires M%128==0, N%64==0, K%64==0 (true for every GEMM in this model).
// Batched over blockIdx.z with (b,h) = (z/Hdiv, z%Hdiv) strides for attention.
// Epilogue fuses: +bias[col], exact GELU, +residual, bf16 mirror store.
// ---------------------------------------------------------------------------
__global__ __launch_bounds__(128)
void gemm_nt_bf16(const bf16* __restrict__ A, const bf16* __restrict__ Bt,
                  float* __restrict__ C, const float* __restrict__ bias,
                  const float* __restrict__ res, bf16* __restrict__ Cbf,
                  int K, int lda, int ldb, int ldc, int Hdiv,
                  long long sAb, long long sAh, long long sBb, long long sBh,
                  long long sCb, long long sCh, int act_gelu)
{
  const int wave = threadIdx.x >> 5;
  const int lane = threadIdx.x & 31;
  const int g    = lane >> 4;        // half-wave
  const int r    = lane & 15;        // row (A) / col (B) within tile
  const int tm   = blockIdx.y * 4 + wave;   // 32-row strip index
  const int tn   = blockIdx.x;              // 64-col strip index
  const int zb   = blockIdx.z / Hdiv;
  const int zh   = blockIdx.z % Hdiv;

  A  += (size_t)(zb * sAb + zh * sAh);
  Bt += (size_t)(zb * sBb + zh * sBh);
  const long long coff = zb * sCb + zh * sCh;

  const bf16* arow0 = A  + (size_t)(tm * 32 + r)      * lda + 16 * g;
  const bf16* arow1 = A  + (size_t)(tm * 32 + 16 + r) * lda + 16 * g;
  const bf16* b0    = Bt + (size_t)(tn * 64 + r)      * ldb + 16 * g;
  const size_t bstep = (size_t)16 * ldb;

  v8f acc[8] = {};

  // ---- ping-pong pipelined k loop (K % 64 == 0), last chunk peeled ----
  // set X fragments (even 32-chunks), set Y fragments (odd 32-chunks)
  v16bf xa0 = ld_afrag(arow0);
  v16bf xa1 = ld_afrag(arow1);
  v16bf xb[4];
#pragma unroll
  for (int j = 0; j < 4; ++j) xb[j] = ld_bfrag(b0 + (size_t)j * bstep);

  int k0 = 0;
  for (; k0 + 64 < K; k0 += 64) {
    __builtin_prefetch(arow0 + k0 + 256, 0, 1);     // 512B ahead on A stream
    // load set Y @ k0+32 while computing on set X
    v16bf ya0 = ld_afrag(arow0 + k0 + 32);
    v16bf ya1 = ld_afrag(arow1 + k0 + 32);
    v16bf yb[4];
#pragma unroll
    for (int j = 0; j < 4; ++j) yb[j] = ld_bfrag(b0 + (size_t)j * bstep + k0 + 32);
#pragma unroll
    for (int j = 0; j < 4; ++j) {
      acc[j]     = wmma_bf16(xa0, xb[j], acc[j]);
      acc[4 + j] = wmma_bf16(xa1, xb[j], acc[4 + j]);
    }
    // load set X @ k0+64 while computing on set Y
    xa0 = ld_afrag(arow0 + k0 + 64);
    xa1 = ld_afrag(arow1 + k0 + 64);
#pragma unroll
    for (int j = 0; j < 4; ++j) xb[j] = ld_bfrag(b0 + (size_t)j * bstep + k0 + 64);
#pragma unroll
    for (int j = 0; j < 4; ++j) {
      acc[j]     = wmma_bf16(ya0, yb[j], acc[j]);
      acc[4 + j] = wmma_bf16(ya1, yb[j], acc[4 + j]);
    }
  }
  // peeled final 64-chunk: k0 = K - 64
  {
    v16bf ya0 = ld_afrag(arow0 + k0 + 32);
    v16bf ya1 = ld_afrag(arow1 + k0 + 32);
    v16bf yb[4];
#pragma unroll
    for (int j = 0; j < 4; ++j) yb[j] = ld_bfrag(b0 + (size_t)j * bstep + k0 + 32);
#pragma unroll
    for (int j = 0; j < 4; ++j) {
      acc[j]     = wmma_bf16(xa0, xb[j], acc[j]);
      acc[4 + j] = wmma_bf16(xa1, xb[j], acc[4 + j]);
    }
#pragma unroll
    for (int j = 0; j < 4; ++j) {
      acc[j]     = wmma_bf16(ya0, yb[j], acc[j]);
      acc[4 + j] = wmma_bf16(ya1, yb[j], acc[4 + j]);
    }
  }

  // ---- epilogue: bias / gelu / residual / fp32 + bf16 mirror stores ----
  float*       Cp = C + coff;
  const float* rp = res ? res + coff : nullptr;
  bf16*        bp = Cbf ? Cbf + coff : nullptr;
#pragma unroll
  for (int s = 0; s < 2; ++s) {
#pragma unroll
    for (int j = 0; j < 4; ++j) {
      const int col = tn * 64 + j * 16 + r;
      const float bv = bias ? bias[col] : 0.0f;
#pragma unroll
      for (int i = 0; i < 8; ++i) {
        const int rr = tm * 32 + s * 16 + i + 8 * g;   // C layout: M = vgpr + 8g
        const size_t idx = (size_t)rr * ldc + col;
        float v = acc[s * 4 + j][i] + bv;
        if (act_gelu) v = 0.5f * v * (1.0f + erff(v * 0.70710678118654752f));
        if (rp) v += rp[idx];
        Cp[idx] = v;
        if (bp) bp[idx] = (bf16)v;
      }
    }
  }
}

// ---------------------------------------------------------------------------
// Embedding + positional encoding (replicates reference's exponent bug:
// theta = pos / 10000^(i/256) with i the even dim index). Writes fp32 + bf16.
// grid*block == NT*Dm exactly.
// ---------------------------------------------------------------------------
__global__ void embed_pe(const int* __restrict__ tok, const float* __restrict__ emb,
                         float* __restrict__ xf, bf16* __restrict__ xb)
{
  const size_t idx = (size_t)blockIdx.x * blockDim.x + threadIdx.x;
  const int row = (int)(idx >> 9);
  const int d   = (int)(idx & 511);
  const int pos = row & (Tn - 1);
  const int i   = d & ~1;
  const float theta = (float)pos * __powf(10000.0f, -(float)i * (1.0f / 256.0f));
  const float pe = (d & 1) ? __cosf(theta) : __sinf(theta);
  const float v = emb[(size_t)tok[row] * Dm + d] * 22.627416997969522f + pe; // sqrt(512)
  xf[idx] = v;
  xb[idx] = (bf16)v;
}

// fp32 [Kd,Nd] -> bf16 transposed [Nd,Kd]
__global__ void conv_wt(const float* __restrict__ W, bf16* __restrict__ Wt,
                        int Kd, int Nd)
{
  const long long i = (long long)blockIdx.x * 256 + threadIdx.x;
  if (i >= (long long)Kd * Nd) return;
  const int k = (int)(i / Nd);
  const int n = (int)(i % Nd);
  Wt[(size_t)n * Kd + k] = (bf16)W[i];
}

// fp32 -> bf16 flat
__global__ void conv_bf(const float* __restrict__ in, bf16* __restrict__ out,
                        long long n)
{
  const long long i = (long long)blockIdx.x * 256 + threadIdx.x;
  if (i < n) out[i] = (bf16)in[i];
}

// V projection [B*T, D] fp32 -> per-head transposed bf16 [B,H,DH,T]
__global__ void trans_heads(const float* __restrict__ v, bf16* __restrict__ vt)
{
  const size_t idx = (size_t)blockIdx.x * blockDim.x + threadIdx.x; // < NT*Dm
  const int row = (int)(idx >> 9);
  const int d   = (int)(idx & 511);
  const int b = row >> 9, t = row & 511;
  const int h = d >> 6,  dh = d & 63;
  vt[((((size_t)b * Hn + h) * DHd + dh) << 9) + t] = (bf16)v[idx];
}

// Row LayerNorm over D=512 (block=256, 2 elems/thread). Optional fp32/bf16 out.
__global__ __launch_bounds__(256)
void layernorm_rows(const float* __restrict__ x, const float* __restrict__ s,
                    const float* __restrict__ bta, float* __restrict__ outf,
                    bf16* __restrict__ outb)
{
  __shared__ float red[256];
  const int row = blockIdx.x, tid = threadIdx.x;
  const float* p = x + (size_t)row * Dm;
  const float a0 = p[tid], a1 = p[tid + 256];
  red[tid] = a0 + a1; __syncthreads();
  for (int st = 128; st; st >>= 1) { if (tid < st) red[tid] += red[tid + st]; __syncthreads(); }
  const float mean = red[0] * (1.0f / Dm);
  __syncthreads();
  const float d0 = a0 - mean, d1 = a1 - mean;
  red[tid] = d0 * d0 + d1 * d1; __syncthreads();
  for (int st = 128; st; st >>= 1) { if (tid < st) red[tid] += red[tid + st]; __syncthreads(); }
  const float inv = rsqrtf(red[0] * (1.0f / Dm) + 1e-5f);
  const float o0 = d0 * inv * s[tid] + bta[tid];
  const float o1 = d1 * inv * s[tid + 256] + bta[tid + 256];
  if (outf) { outf[(size_t)row * Dm + tid] = o0; outf[(size_t)row * Dm + tid + 256] = o1; }
  if (outb) { outb[(size_t)row * Dm + tid] = (bf16)o0; outb[(size_t)row * Dm + tid + 256] = (bf16)o1; }
}

// In-place softmax over rows of length T=512 (block=256, 2 elems/thread).
// Applies 1/sqrt(DH) scale and optional causal mask; writes fp32 back in place
// (the returned attention-weight tensors) + a bf16 copy for the P*V GEMM.
__global__ __launch_bounds__(256)
void softmax_rows(float* __restrict__ dat, bf16* __restrict__ outb,
                  int causal, float scale)
{
  __shared__ float red[256];
  const int row = blockIdx.x;
  const int tq  = row & (Tn - 1);
  const int tid = threadIdx.x;
  float* p = dat + (size_t)row * Tn;
  const int limit = causal ? (tq + 1) : Tn;
  const float v0 = p[tid] * scale, v1 = p[tid + 256] * scale;
  const bool m0 = tid < limit, m1 = (tid + 256) < limit;
  const float NEG = -3.4028235e38f;
  red[tid] = fmaxf(m0 ? v0 : NEG, m1 ? v1 : NEG); __syncthreads();
  for (int st = 128; st; st >>= 1) { if (tid < st) red[tid] = fmaxf(red[tid], red[tid + st]); __syncthreads(); }
  const float mx = red[0];
  __syncthreads();
  const float e0 = m0 ? __expf(v0 - mx) : 0.0f;
  const float e1 = m1 ? __expf(v1 - mx) : 0.0f;
  red[tid] = e0 + e1; __syncthreads();
  for (int st = 128; st; st >>= 1) { if (tid < st) red[tid] += red[tid + st]; __syncthreads(); }
  const float inv = 1.0f / red[0];
  const float r0 = e0 * inv, r1 = e1 * inv;
  p[tid] = r0; p[tid + 256] = r1;
  if (outb) {
    outb[(size_t)row * Tn + tid] = (bf16)r0;
    outb[(size_t)row * Tn + tid + 256] = (bf16)r1;
  }
}

// ---------------------------------------------------------------------------
// Host orchestration
// ---------------------------------------------------------------------------
extern "C" void kernel_launch(void* const* d_in, const int* in_sizes, int n_in,
                              void* d_out, int out_size, void* d_ws, size_t ws_size,
                              hipStream_t stream)
{
  (void)in_sizes; (void)n_in; (void)out_size; (void)ws_size;

  // input order: src, tgt, then params dict depth-first in insertion order
  const int*   src     = (const int*)d_in[0];
  const int*   tgt     = (const int*)d_in[1];
  const float* src_emb = (const float*)d_in[2];
  const float* tgt_emb = (const float*)d_in[3];
  // encoder
  const float* eWq = (const float*)d_in[4];  const float* ebq = (const float*)d_in[5];
  const float* eWk = (const float*)d_in[6];  const float* ebk = (const float*)d_in[7];
  const float* eWv = (const float*)d_in[8];  const float* ebv = (const float*)d_in[9];
  const float* eWo = (const float*)d_in[10]; const float* ebo = (const float*)d_in[11];
  const float* eln1s = (const float*)d_in[12]; const float* eln1b = (const float*)d_in[13];
  const float* eln2s = (const float*)d_in[14]; const float* eln2b = (const float*)d_in[15];
  const float* eW1 = (const float*)d_in[16]; const float* eb1 = (const float*)d_in[17];
  const float* eW2 = (const float*)d_in[18]; const float* eb2 = (const float*)d_in[19];
  const float* enorm_s = (const float*)d_in[20]; const float* enorm_b = (const float*)d_in[21];
  // decoder
  const float* sWq = (const float*)d_in[22]; const float* sbq = (const float*)d_in[23];
  const float* sWk = (const float*)d_in[24]; const float* sbk = (const float*)d_in[25];
  const float* sWv = (const float*)d_in[26]; const float* sbv = (const float*)d_in[27];
  const float* sWo = (const float*)d_in[28]; const float* sbo = (const float*)d_in[29];
  const float* cWq = (const float*)d_in[30]; const float* cbq = (const float*)d_in[31];
  const float* cWk = (const float*)d_in[32]; const float* cbk = (const float*)d_in[33];
  const float* cWv = (const float*)d_in[34]; const float* cbv = (const float*)d_in[35];
  const float* cWo = (const float*)d_in[36]; const float* cbo = (const float*)d_in[37];
  const float* lns_s = (const float*)d_in[38]; const float* lns_b = (const float*)d_in[39];
  const float* lnc_s = (const float*)d_in[40]; const float* lnc_b = (const float*)d_in[41];
  const float* lnf_s = (const float*)d_in[42]; const float* lnf_b = (const float*)d_in[43];
  const float* dW1 = (const float*)d_in[44]; const float* db1 = (const float*)d_in[45];
  const float* dW2 = (const float*)d_in[46]; const float* db2 = (const float*)d_in[47];
  const float* Wout = (const float*)d_in[48]; const float* bout = (const float*)d_in[49];

  // output slices: logits | self_ws | cross_ws
  float* out_logits = (float*)d_out;
  float* out_self   = out_logits + (size_t)NT * Vv;                     // B*T*V
  float* out_cross  = out_self + (size_t)LDn * Bn * Hn * Tn * Tn;

  // workspace bump allocator (~230 MB total)
  char* wsbase = (char*)d_ws;
  size_t cur = 0;
  auto allocF = [&](size_t n) { float* p = (float*)(wsbase + cur); cur += (n * 4 + 255) & ~(size_t)255; return p; };
  auto allocH = [&](size_t n) { bf16*  p = (bf16*)(wsbase + cur);  cur += (n * 2 + 255) & ~(size_t)255; return p; };

  // transposed bf16 weights
  bf16 *eWqt = allocH((size_t)LEn * Dm * Dm), *eWkt = allocH((size_t)LEn * Dm * Dm);
  bf16 *eWvt = allocH((size_t)LEn * Dm * Dm), *eWot = allocH((size_t)LEn * Dm * Dm);
  bf16 *eW1t = allocH((size_t)LEn * Dm * Fff), *eW2t = allocH((size_t)LEn * Dm * Fff);
  bf16 *sWqt = allocH((size_t)LDn * Dm * Dm), *sWkt = allocH((size_t)LDn * Dm * Dm);
  bf16 *sWvt = allocH((size_t)LDn * Dm * Dm), *sWot = allocH((size_t)LDn * Dm * Dm);
  bf16 *cWqt = allocH((size_t)LDn * Dm * Dm), *cWkt = allocH((size_t)LDn * Dm * Dm);
  bf16 *cWvt = allocH((size_t)LDn * Dm * Dm), *cWot = allocH((size_t)LDn * Dm * Dm);
  bf16 *dW1t = allocH((size_t)LDn * Dm * Fff), *dW2t = allocH((size_t)LDn * Dm * Fff);
  bf16 *Woutt = allocH((size_t)Vv * Dm);
  // activations
  float* scoresF = allocF((size_t)Bn * Hn * Tn * Tn);   // encoder attn logits
  bf16*  scoresB = allocH((size_t)Bn * Hn * Tn * Tn);   // softmax probs (bf16)
  float* tmpF = allocF((size_t)NT * Fff);               // fp32 dump / ffn hidden
  bf16*  hB   = allocH((size_t)NT * Fff);               // gelu hidden (bf16)
  float* xF  = allocF((size_t)NT * Dm);  bf16* xB  = allocH((size_t)NT * Dm);
  float* t2F = allocF((size_t)NT * Dm);
  float* yF  = allocF((size_t)NT * Dm);  bf16* yB  = allocH((size_t)NT * Dm);
  float* vF  = allocF((size_t)NT * Dm);  bf16* vtB = allocH((size_t)NT * Dm);
  float* ctxF = allocF((size_t)NT * Dm); bf16* ctxB = allocH((size_t)NT * Dm);
  bf16* ynB  = allocH((size_t)NT * Dm);
  bf16* qB   = allocH((size_t)NT * Dm);
  bf16* kB   = allocH((size_t)NT * Dm);
  bf16* encB = allocH((size_t)NT * Dm);

  // ---- weight conversion (fp32 -> transposed bf16) ----
  auto wt = [&](const float* W, bf16* Wt, int Kd, int Nd) {
    long long n = (long long)Kd * Nd;
    conv_wt<<<(unsigned)((n + 255) / 256), 256, 0, stream>>>(W, Wt, Kd, Nd);
  };
  for (int l = 0; l < LEn; ++l) {
    size_t o = (size_t)l * Dm * Dm, of = (size_t)l * Dm * Fff;
    wt(eWq + o, eWqt + o, Dm, Dm);  wt(eWk + o, eWkt + o, Dm, Dm);
    wt(eWv + o, eWvt + o, Dm, Dm);  wt(eWo + o, eWot + o, Dm, Dm);
    wt(eW1 + of, eW1t + of, Dm, Fff);  wt(eW2 + of, eW2t + of, Fff, Dm);
  }
  for (int l = 0; l < LDn; ++l) {
    size_t o = (size_t)l * Dm * Dm, of = (size_t)l * Dm * Fff;
    wt(sWq + o, sWqt + o, Dm, Dm);  wt(sWk + o, sWkt + o, Dm, Dm);
    wt(sWv + o, sWvt + o, Dm, Dm);  wt(sWo + o, sWot + o, Dm, Dm);
    wt(cWq + o, cWqt + o, Dm, Dm);  wt(cWk + o, cWkt + o, Dm, Dm);
    wt(cWv + o, cWvt + o, Dm, Dm);  wt(cWo + o, cWot + o, Dm, Dm);
    wt(dW1 + of, dW1t + of, Dm, Fff);  wt(dW2 + of, dW2t + of, Fff, Dm);
  }
  wt(Wout, Woutt, Dm, Vv);

  // GEMM launcher (NT, batched over z with (b,h) strides); block = 128x64 tile
  auto gemm = [&](const bf16* A, const bf16* Bt, float* C,
                  const float* bias, const float* res, bf16* Cbf,
                  int M, int N, int K, int lda, int ldb, int ldc,
                  int nz, int Hdiv,
                  long long sAb, long long sAh, long long sBb, long long sBh,
                  long long sCb, long long sCh, int gelu) {
    dim3 grid(N / 64, M / 128, nz);
    gemm_nt_bf16<<<grid, 128, 0, stream>>>(A, Bt, C, bias, res, Cbf,
        K, lda, ldb, ldc, Hdiv, sAb, sAh, sBb, sBh, sCb, sCh, gelu);
  };

  const long long TD  = (long long)Tn * Dm;       // 262144
  const long long TT  = (long long)Tn * Tn;       // 262144
  const long long HTT = (long long)Hn * TT;       // per-batch stride in [B,H,T,T]
  const long long HDT = (long long)Hn * DHd * Tn; // per-batch stride in [B,H,DH,T]
  const long long DT  = (long long)DHd * Tn;      // 32768
  const int ebl = (NT * Dm) / 256;                // 4096 blocks for elementwise

  // ---------------- encoder ----------------
  embed_pe<<<ebl, 256, 0, stream>>>(src, src_emb, xF, xB);
  for (int l = 0; l < LEn; ++l) {
    const size_t wo = (size_t)l * Dm * Dm, wf = (size_t)l * Dm * Fff;
    gemm(xB, eWqt + wo, tmpF, ebq + (size_t)l * Dm, nullptr, qB,
         NT, Dm, Dm, Dm, Dm, Dm, 1, 1, 0,0,0,0,0,0, 0);
    gemm(xB, eWkt + wo, tmpF, ebk + (size_t)l * Dm, nullptr, kB,
         NT, Dm, Dm, Dm, Dm, Dm, 1, 1, 0,0,0,0,0,0, 0);
    gemm(xB, eWvt + wo, vF, ebv + (size_t)l * Dm, nullptr, nullptr,
         NT, Dm, Dm, Dm, Dm, Dm, 1, 1, 0,0,0,0,0,0, 0);
    trans_heads<<<ebl, 256, 0, stream>>>(vF, vtB);
    // S = Q * K^T  (per b,h)
    gemm(qB, kB, scoresF, nullptr, nullptr, nullptr,
         Tn, Tn, DHd, Dm, Dm, Tn, Bn * Hn, Hn, TD, DHd, TD, DHd, HTT, TT, 0);
    softmax_rows<<<Bn * Hn * Tn, 256, 0, stream>>>(scoresF, scoresB, 0, 0.125f);
    // ctx = P * V
    gemm(scoresB, vtB, ctxF, nullptr, nullptr, ctxB,
         Tn, DHd, Tn, Tn, Tn, Dm, Bn * Hn, Hn, HTT, TT, HDT, DT, TD, DHd, 0);
    // x + attn_out ; post-norm
    gemm(ctxB, eWot + wo, t2F, ebo + (size_t)l * Dm, xF, nullptr,
         NT, Dm, Dm, Dm, Dm, Dm, 1, 1, 0,0,0,0,0,0, 0);
    layernorm_rows<<<NT, 256, 0, stream>>>(t2F, eln1s + (size_t)l * Dm, eln1b + (size_t)l * Dm, xF, xB);
    // FFN
    gemm(xB, eW1t + wf, tmpF, eb1 + (size_t)l * Fff, nullptr, hB,
         NT, Fff, Dm, Dm, Dm, Fff, 1, 1, 0,0,0,0,0,0, 1 /*gelu*/);
    gemm(hB, eW2t + wf, t2F, eb2 + (size_t)l * Dm, xF, nullptr,
         NT, Dm, Fff, Fff, Fff, Dm, 1, 1, 0,0,0,0,0,0, 0);
    layernorm_rows<<<NT, 256, 0, stream>>>(t2F, eln2s + (size_t)l * Dm, eln2b + (size_t)l * Dm, xF, xB);
  }
  layernorm_rows<<<NT, 256, 0, stream>>>(xF, enorm_s, enorm_b, nullptr, encB);

  // ---------------- decoder (pre-norm) ----------------
  embed_pe<<<ebl, 256, 0, stream>>>(tgt, tgt_emb, yF, yB);
  for (int l = 0; l < LDn; ++l) {
    const size_t wo = (size_t)l * Dm * Dm, wf = (size_t)l * Dm * Fff;
    // -- causal self-attention --
    layernorm_rows<<<NT, 256, 0, stream>>>(yF, lns_s + (size_t)l * Dm, lns_b + (size_t)l * Dm, nullptr, ynB);
    gemm(ynB, sWqt + wo, tmpF, sbq + (size_t)l * Dm, nullptr, qB, NT, Dm, Dm, Dm, Dm, Dm, 1,1, 0,0,0,0,0,0, 0);
    gemm(ynB, sWkt + wo, tmpF, sbk + (size_t)l * Dm, nullptr, kB, NT, Dm, Dm, Dm, Dm, Dm, 1,1, 0,0,0,0,0,0, 0);
    gemm(ynB, sWvt + wo, vF,   sbv + (size_t)l * Dm, nullptr, nullptr, NT, Dm, Dm, Dm, Dm, Dm, 1,1, 0,0,0,0,0,0, 0);
    trans_heads<<<ebl, 256, 0, stream>>>(vF, vtB);
    float* selfC = out_self + (size_t)l * Bn * Hn * Tn * Tn;  // probs land in d_out
    gemm(qB, kB, selfC, nullptr, nullptr, nullptr,
         Tn, Tn, DHd, Dm, Dm, Tn, Bn * Hn, Hn, TD, DHd, TD, DHd, HTT, TT, 0);
    softmax_rows<<<Bn * Hn * Tn, 256, 0, stream>>>(selfC, scoresB, 1 /*causal*/, 0.125f);
    gemm(scoresB, vtB, ctxF, nullptr, nullptr, ctxB,
         Tn, DHd, Tn, Tn, Tn, Dm, Bn * Hn, Hn, HTT, TT, HDT, DT, TD, DHd, 0);
    gemm(ctxB, sWot + wo, yF, sbo + (size_t)l * Dm, yF /*residual in-place*/, nullptr,
         NT, Dm, Dm, Dm, Dm, Dm, 1,1, 0,0,0,0,0,0, 0);
    // -- cross-attention --
    layernorm_rows<<<NT, 256, 0, stream>>>(yF, lnc_s + (size_t)l * Dm, lnc_b + (size_t)l * Dm, nullptr, ynB);
    gemm(ynB,  cWqt + wo, tmpF, cbq + (size_t)l * Dm, nullptr, qB, NT, Dm, Dm, Dm, Dm, Dm, 1,1, 0,0,0,0,0,0, 0);
    gemm(encB, cWkt + wo, tmpF, cbk + (size_t)l * Dm, nullptr, kB, NT, Dm, Dm, Dm, Dm, Dm, 1,1, 0,0,0,0,0,0, 0);
    gemm(encB, cWvt + wo, vF,   cbv + (size_t)l * Dm, nullptr, nullptr, NT, Dm, Dm, Dm, Dm, Dm, 1,1, 0,0,0,0,0,0, 0);
    trans_heads<<<ebl, 256, 0, stream>>>(vF, vtB);
    float* crossC = out_cross + (size_t)l * Bn * Hn * Tn * Tn;
    gemm(qB, kB, crossC, nullptr, nullptr, nullptr,
         Tn, Tn, DHd, Dm, Dm, Tn, Bn * Hn, Hn, TD, DHd, TD, DHd, HTT, TT, 0);
    softmax_rows<<<Bn * Hn * Tn, 256, 0, stream>>>(crossC, scoresB, 0, 0.125f);
    gemm(scoresB, vtB, ctxF, nullptr, nullptr, ctxB,
         Tn, DHd, Tn, Tn, Tn, Dm, Bn * Hn, Hn, HTT, TT, HDT, DT, TD, DHd, 0);
    gemm(ctxB, cWot + wo, yF, cbo + (size_t)l * Dm, yF, nullptr,
         NT, Dm, Dm, Dm, Dm, Dm, 1,1, 0,0,0,0,0,0, 0);
    // -- FFN --
    layernorm_rows<<<NT, 256, 0, stream>>>(yF, lnf_s + (size_t)l * Dm, lnf_b + (size_t)l * Dm, nullptr, ynB);
    gemm(ynB, dW1t + wf, tmpF, db1 + (size_t)l * Fff, nullptr, hB,
         NT, Fff, Dm, Dm, Dm, Fff, 1,1, 0,0,0,0,0,0, 1 /*gelu*/);
    gemm(hB, dW2t + wf, yF, db2 + (size_t)l * Dm, yF, nullptr,
         NT, Dm, Fff, Fff, Fff, Dm, 1,1, 0,0,0,0,0,0, 0);
  }

  // ---------------- logits ----------------
  conv_bf<<<ebl, 256, 0, stream>>>(yF, yB, (long long)NT * Dm);
  gemm(yB, Woutt, out_logits, bout, nullptr, nullptr,
       NT, Vv, Dm, Dm, Dm, Vv, 1, 1, 0,0,0,0,0,0, 0);
}
